// HAN_38010460570177
// MI455X (gfx1250) — matrix operand order repaired and checked by reading, not compiled, not run on hardware
//
#include <hip/hip_runtime.h>
#include <hip/hip_bf16.h>
#include <math.h>

#define NA_N    50000
#define NP_N    100000
#define E_AP_C  400000
#define E_PA_C  400000
#define E_AA_C  200000
#define NEG_SLOPE_C 0.2f

typedef __attribute__((ext_vector_type(16))) __bf16 v16bf;
typedef __attribute__((ext_vector_type(2)))  __bf16 bf16x2;
typedef __attribute__((ext_vector_type(8)))  float  v8f;
typedef __attribute__((ext_vector_type(8)))  unsigned short ushort8;
typedef __attribute__((ext_vector_type(2)))  unsigned int   uint2v;

union FragBF {
    v16bf   bf;
    ushort8 us[2];
};

// native v_cvt_bf16_f32 / v_cvt_pk_bf16_f32 (RNE) via __bf16 casts
__device__ __forceinline__ unsigned short f2bf(float x) {
    __bf16 b = (__bf16)x;
    return __builtin_bit_cast(unsigned short, b);
}
__device__ __forceinline__ unsigned int pack2bf(float lo, float hi) {
    bf16x2 v = { (__bf16)lo, (__bf16)hi };
    return __builtin_bit_cast(unsigned int, v);
}

#define MODE_STORE 0
#define MODE_SCORE 2

// C[M,N] = A[M,K] @ B[K,N] (+bias) with bf16 WMMA.
// mode STORE: C = acc + bias[n]
// mode SCORE: atomicAdd(score, sum over tile of tanh(acc + bias[n]) * q[n])
__global__ __launch_bounds__(256)
void gemm_bf16(const float* __restrict__ A, const float* __restrict__ B,
               const float* __restrict__ bias, float* __restrict__ C,
               int M, int N, int K, int mode,
               const float* __restrict__ q, float* __restrict__ score)
{
    __shared__ unsigned int   AsU[128 * 16];   // 128 rows x 32 bf16 (16 uints/row)
    __shared__ unsigned short Bs[64 * 32];     // transposed: 64 cols x 32 bf16
    __shared__ float red[256];

    const int tid  = threadIdx.x;
    const int lane = tid & 31;
    const int wave = tid >> 5;
    const int wm   = wave >> 1;          // 0..3
    const int wn   = wave & 1;           // 0..1
    const int bm   = blockIdx.y * 128;
    const int bn   = blockIdx.x * 64;
    const bool fullM = (bm + 128 <= M);

    v8f acc[2][2];
    v8f zero = {0.f, 0.f, 0.f, 0.f, 0.f, 0.f, 0.f, 0.f};
    acc[0][0] = zero; acc[0][1] = zero; acc[1][0] = zero; acc[1][1] = zero;

    const int kbA = (lane >= 16) ? 8  : 0;   // ISA 16-bit A layout
    const int kbB = (lane >= 16) ? 8  : 0;   // B: same 16-bit layout as A
    const unsigned short* AsS = (const unsigned short*)AsU;
    const int arow = wm * 32 + (lane & 15);
    const int bcol = wn * 32 + (lane & 15);

    for (int k0 = 0; k0 < K; k0 += 32) {
        // ---- stage A tile 128x32 (fp32 -> bf16), float4 loads, b64 LDS stores ----
        if (fullM) {
            #pragma unroll
            for (int i = 0; i < 4; ++i) {
                int idx4 = (i * 256 + tid) * 4;
                int r = idx4 >> 5, c = idx4 & 31;
                const float4 v = *(const float4*)(A + (size_t)(bm + r) * K + k0 + c);
                uint2v p; p.x = pack2bf(v.x, v.y); p.y = pack2bf(v.z, v.w);
                *(uint2v*)&AsU[r * 16 + (c >> 1)] = p;
            }
        } else {
            #pragma unroll
            for (int i = 0; i < 4; ++i) {
                int idx4 = (i * 256 + tid) * 4;
                int r = idx4 >> 5, c = idx4 & 31;
                int gr = bm + r;
                float4 v = make_float4(0.f, 0.f, 0.f, 0.f);
                if (gr < M) v = *(const float4*)(A + (size_t)gr * K + k0 + c);
                uint2v p; p.x = pack2bf(v.x, v.y); p.y = pack2bf(v.z, v.w);
                *(uint2v*)&AsU[r * 16 + (c >> 1)] = p;
            }
        }
        // ---- stage B tile 32x64, transposed to [n][k] ----
        #pragma unroll
        for (int i = 0; i < 2; ++i) {
            int idx4 = (i * 256 + tid) * 4;
            int kk = idx4 >> 6, n = idx4 & 63;
            const float4 v = *(const float4*)(B + (size_t)(k0 + kk) * N + bn + n);
            Bs[(n + 0) * 32 + kk] = f2bf(v.x);
            Bs[(n + 1) * 32 + kk] = f2bf(v.y);
            Bs[(n + 2) * 32 + kk] = f2bf(v.z);
            Bs[(n + 3) * 32 + kk] = f2bf(v.w);
        }
        __syncthreads();

        FragBF a[2], b[2];
        #pragma unroll
        for (int mi = 0; mi < 2; ++mi) {
            int row = arow + mi * 16;
            a[mi].us[0] = *(const ushort8*)&AsS[row * 32 + kbA];
            a[mi].us[1] = *(const ushort8*)&AsS[row * 32 + 16 + kbA];
        }
        #pragma unroll
        for (int ni = 0; ni < 2; ++ni) {
            int col = bcol + ni * 16;
            b[ni].us[0] = *(const ushort8*)&Bs[col * 32 + kbB];
            b[ni].us[1] = *(const ushort8*)&Bs[col * 32 + kbB + 16];
        }

        #pragma unroll
        for (int mi = 0; mi < 2; ++mi)
            #pragma unroll
            for (int ni = 0; ni < 2; ++ni)
                acc[mi][ni] = __builtin_amdgcn_wmma_f32_16x16x32_bf16(
                    false, a[mi].bf, false, b[ni].bf, (short)0, acc[mi][ni],
                    false, false);
        __syncthreads();
    }

    const int rofs = (lane >= 16) ? 8 : 0;
    if (mode == MODE_STORE) {
        #pragma unroll
        for (int mi = 0; mi < 2; ++mi)
            #pragma unroll
            for (int ni = 0; ni < 2; ++ni) {
                int col = bn + wn * 32 + ni * 16 + (lane & 15);
                float bv = bias[col];
                #pragma unroll
                for (int r = 0; r < 8; ++r) {
                    int row = bm + wm * 32 + mi * 16 + r + rofs;
                    if (row < M) C[(size_t)row * N + col] = acc[mi][ni][r] + bv;
                }
            }
    } else {
        float part = 0.f;
        #pragma unroll
        for (int mi = 0; mi < 2; ++mi)
            #pragma unroll
            for (int ni = 0; ni < 2; ++ni) {
                int col = bn + wn * 32 + ni * 16 + (lane & 15);
                float bv = bias[col], qv = q[col];
                #pragma unroll
                for (int r = 0; r < 8; ++r) {
                    int row = bm + wm * 32 + mi * 16 + r + rofs;
                    if (row < M) part += tanhf(acc[mi][ni][r] + bv) * qv;
                }
            }
        red[tid] = part;
        __syncthreads();
        for (int s = 128; s > 0; s >>= 1) {
            if (tid < s) red[tid] += red[tid + s];
            __syncthreads();
        }
        if (tid == 0) atomicAdd(score, red[0]);
    }
}

// s[i] = h[i,:] . a    (one wave per row, float4 per lane)
__global__ __launch_bounds__(256)
void rowdot(const float* __restrict__ h, const float* __restrict__ a,
            float* __restrict__ s, int n, int D)
{
    int row  = blockIdx.x * 8 + (threadIdx.x >> 5);
    int lane = threadIdx.x & 31;
    if (row >= n) return;
    float p = 0.f;
    for (int d = lane * 4; d < D; d += 128) {
        float4 v  = *(const float4*)&h[(size_t)row * D + d];
        float4 av = *(const float4*)&a[d];
        p += v.x * av.x + v.y * av.y + v.z * av.z + v.w * av.w;
    }
    #pragma unroll
    for (int off = 16; off > 0; off >>= 1) p += __shfl_xor(p, off, 32);
    if (lane == 0) s[row] = p;
}

__global__ void fill_f32(float* __restrict__ p, float v, size_t n) {
    size_t i = (size_t)blockIdx.x * blockDim.x + threadIdx.x;
    if (i < n) p[i] = v;
}

__global__ void relu4_f32(float4* __restrict__ p, size_t n4) {
    size_t i = (size_t)blockIdx.x * blockDim.x + threadIdx.x;
    if (i < n4) {
        float4 v = p[i];
        v.x = fmaxf(v.x, 0.f); v.y = fmaxf(v.y, 0.f);
        v.z = fmaxf(v.z, 0.f); v.w = fmaxf(v.w, 0.f);
        p[i] = v;
    }
}

__device__ __forceinline__ void atomicMaxF(float* addr, float val) {
    if (val >= 0.f) atomicMax((int*)addr, __float_as_int(val));
    else            atomicMin((unsigned int*)addr, __float_as_uint(val));
}

__global__ void edge_alpha(const int* __restrict__ src, const int* __restrict__ dst,
                           const float* __restrict__ ssrc, const float* __restrict__ sdst,
                           float* __restrict__ alpha, float* __restrict__ m, int E)
{
    int e = blockIdx.x * blockDim.x + threadIdx.x;
    if (e >= E) return;
    float v = ssrc[src[e]] + sdst[dst[e]];
    v = (v > 0.f) ? v : NEG_SLOPE_C * v;
    alpha[e] = v;
    atomicMaxF(&m[dst[e]], v);
}

__global__ void edge_exp(const int* __restrict__ dst, const float* __restrict__ alpha,
                         const float* __restrict__ m, float* __restrict__ ex,
                         float* __restrict__ denom, int E)
{
    int e = blockIdx.x * blockDim.x + threadIdx.x;
    if (e >= E) return;
    int t = dst[e];
    float v = __expf(alpha[e] - m[t]);
    ex[e] = v;
    atomicAdd(&denom[t], v);
}

// one thread = 4 channels of one edge (float4 gather, 4 atomic adds)
__global__ void edge_scatter4(const int* __restrict__ src, const int* __restrict__ dst,
                              const float* __restrict__ ex, const float* __restrict__ denom,
                              const float* __restrict__ h, float* __restrict__ out,
                              long long tot4, int D)
{
    long long f = (long long)blockIdx.x * blockDim.x + threadIdx.x;
    if (f >= tot4) return;
    int Dq = D >> 2;
    int e  = (int)(f / Dq);
    int d  = (int)(f - (long long)e * Dq) << 2;
    int s = src[e], t = dst[e];
    float w = ex[e] / denom[t];
    float4 hv = *(const float4*)&h[(size_t)s * D + d];
    float* o = &out[(size_t)t * D + d];
    atomicAdd(o + 0, hv.x * w);
    atomicAdd(o + 1, hv.y * w);
    atomicAdd(o + 2, hv.z * w);
    atomicAdd(o + 3, hv.w * w);
}

__global__ void softmax2(const float* __restrict__ scores, float invN,
                         float* __restrict__ attn)
{
    float s0 = scores[0] * invN, s1 = scores[1] * invN;
    float mx = fmaxf(s0, s1);
    float e0 = __expf(s0 - mx), e1 = __expf(s1 - mx);
    float d = e0 + e1;
    attn[0] = e0 / d;
    attn[1] = e1 / d;
}

__global__ void combine2_v4(const float4* __restrict__ x0, const float4* __restrict__ x1,
                            const float* __restrict__ attn, float4* __restrict__ out, size_t n4)
{
    size_t i = (size_t)blockIdx.x * blockDim.x + threadIdx.x;
    if (i >= n4) return;
    float a0 = attn[0], a1 = attn[1];
    float4 u = x0[i], v = x1[i], r;
    r.x = a0 * u.x + a1 * v.x;
    r.y = a0 * u.y + a1 * v.y;
    r.z = a0 * u.z + a1 * v.z;
    r.w = a0 * u.w + a1 * v.w;
    out[i] = r;
}

extern "C" void kernel_launch(void* const* d_in, const int* in_sizes, int n_in,
                              void* d_out, int out_size, void* d_ws, size_t ws_size,
                              hipStream_t stream)
{
    const float* x_a  = (const float*)d_in[0];
    const float* x_p  = (const float*)d_in[1];
    const float* W1   = (const float*)d_in[2];
    const float* b1   = (const float*)d_in[3];
    const float* att1 = (const float*)d_in[4];
    const float* kW1  = (const float*)d_in[5];
    const float* kb1  = (const float*)d_in[6];
    const float* q1   = (const float*)d_in[7];
    const float* W2   = (const float*)d_in[8];
    const float* b2   = (const float*)d_in[9];
    const float* att2 = (const float*)d_in[10];
    const float* kW2  = (const float*)d_in[11];
    const float* kb2  = (const float*)d_in[12];
    const float* q2   = (const float*)d_in[13];
    const int*   e_ap = (const int*)d_in[14];
    const int*   e_pa = (const int*)d_in[15];
    const int*   e_aa = (const int*)d_in[16];

    // workspace layout (256B aligned)
    size_t off = 0;
    auto walloc = [&](size_t bytes) -> float* {
        float* p = (float*)((char*)d_ws + off);
        off += (bytes + 255) & ~(size_t)255;
        return p;
    };
    float* h_a    = walloc((size_t)NA_N * 512 * 4);
    float* h_p    = walloc((size_t)NP_N * 512 * 4);
    float* out_p  = walloc((size_t)NP_N * 512 * 4);
    float* out_a1 = walloc((size_t)NA_N * 512 * 4);
    float* out_a2 = walloc((size_t)NA_N * 512 * 4);
    float* ssrc   = walloc((size_t)NP_N * 4);
    float* sdst   = walloc((size_t)NP_N * 4);
    float* mbuf   = walloc((size_t)NP_N * 4);
    float* denom  = walloc((size_t)NP_N * 4);
    float* alpha  = walloc((size_t)E_AP_C * 4);
    float* exv    = walloc((size_t)E_AP_C * 4);
    float* scores = walloc(2 * 4);
    float* attnw  = walloc(2 * 4);

    auto fill = [&](float* p, float v, size_t n) {
        fill_f32<<<dim3((unsigned)((n + 255) / 256)), 256, 0, stream>>>(p, v, n);
    };
    auto gemm = [&](const float* A, const float* B, const float* bias, float* C,
                    int M, int N, int K, int mode, const float* q, float* sc) {
        dim3 g((unsigned)(N / 64), (unsigned)((M + 127) / 128));
        gemm_bf16<<<g, 256, 0, stream>>>(A, B, bias, C, M, N, K, mode, q, sc);
    };
    auto edge_attn = [&](const float* hsrc, int nsrc, const float* hdst, int ndst,
                         const float* a_src, const float* a_dst,
                         const int* esrc, const int* edst, int E,
                         float* outbuf, int D) {
        rowdot<<<dim3((unsigned)((nsrc + 7) / 8)), 256, 0, stream>>>(hsrc, a_src, ssrc, nsrc, D);
        rowdot<<<dim3((unsigned)((ndst + 7) / 8)), 256, 0, stream>>>(hdst, a_dst, sdst, ndst, D);
        fill(mbuf, -INFINITY, (size_t)ndst);
        fill(denom, 0.f, (size_t)ndst);
        fill(outbuf, 0.f, (size_t)ndst * D);
        edge_alpha<<<dim3((unsigned)((E + 255) / 256)), 256, 0, stream>>>(esrc, edst, ssrc, sdst, alpha, mbuf, E);
        edge_exp<<<dim3((unsigned)((E + 255) / 256)), 256, 0, stream>>>(edst, alpha, mbuf, exv, denom, E);
        long long tot4 = (long long)E * (D / 4);
        edge_scatter4<<<dim3((unsigned)((tot4 + 255) / 256)), 256, 0, stream>>>(esrc, edst, exv, denom, hsrc, outbuf, tot4, D);
        size_t n4 = (size_t)ndst * D / 4;
        relu4_f32<<<dim3((unsigned)((n4 + 255) / 256)), 256, 0, stream>>>((float4*)outbuf, n4);
    };

    // ---------------- Layer 1 (512 -> 512) ----------------
    gemm(x_a, W1,             b1,       h_a, NA_N, 512, 512, MODE_STORE, nullptr, nullptr);
    gemm(x_p, W1 + 512 * 512, b1 + 512, h_p, NP_N, 512, 512, MODE_STORE, nullptr, nullptr);

    edge_attn(h_a, NA_N, h_p, NP_N, att1 + 0 * 512, att1 + 1 * 512,
              e_ap, e_ap + E_AP_C, E_AP_C, out_p, 512);
    edge_attn(h_p, NP_N, h_a, NA_N, att1 + 2 * 512, att1 + 3 * 512,
              e_pa, e_pa + E_PA_C, E_PA_C, out_a1, 512);
    edge_attn(h_a, NA_N, h_a, NA_N, att1 + 4 * 512, att1 + 5 * 512,
              e_aa, e_aa + E_AA_C, E_AA_C, out_a2, 512);

    fill(scores, 0.f, 2);
    gemm(out_a1, kW1, kb1, nullptr, NA_N, 512, 512, MODE_SCORE, q1, scores + 0);
    gemm(out_a2, kW1, kb1, nullptr, NA_N, 512, 512, MODE_SCORE, q1, scores + 1);
    softmax2<<<1, 1, 0, stream>>>(scores, 1.f / (float)NA_N, attnw);
    combine2_v4<<<dim3((unsigned)(((size_t)NA_N * 128 + 255) / 256)), 256, 0, stream>>>(
        (const float4*)out_a1, (const float4*)out_a2, attnw, (float4*)h_a, (size_t)NA_N * 128);
    // paper semantic attention over a single metapath == identity -> out_p as-is.

    // ---------------- Layer 2 (512 -> 256) ----------------
    float* x_a2 = h_a;          // [NA, 512]
    float* x_p2 = out_p;        // [NP, 512]
    float* h_a2 = out_a1;       // reuse: [NA, 256]
    float* h_p2 = out_a2;       // reuse: [NP, 256] (fits: NA*512 == NP*256)
    float* oa1  = h_p;                          // [NA, 256]
    float* oa2  = h_p + (size_t)NA_N * 256;     // [NA, 256]

    gemm(x_a2, W2,             b2,       h_a2, NA_N, 256, 512, MODE_STORE, nullptr, nullptr);
    gemm(x_p2, W2 + 512 * 256, b2 + 256, h_p2, NP_N, 256, 512, MODE_STORE, nullptr, nullptr);

    // layer-2 paper aggregation (e_ap) is dead code: only h_a is returned. Skip it.
    edge_attn(h_p2, NP_N, h_a2, NA_N, att2 + 2 * 256, att2 + 3 * 256,
              e_pa, e_pa + E_PA_C, E_PA_C, oa1, 256);
    edge_attn(h_a2, NA_N, h_a2, NA_N, att2 + 4 * 256, att2 + 5 * 256,
              e_aa, e_aa + E_AA_C, E_AA_C, oa2, 256);

    fill(scores, 0.f, 2);
    gemm(oa1, kW2, kb2, nullptr, NA_N, 256, 256, MODE_SCORE, q2, scores + 0);
    gemm(oa2, kW2, kb2, nullptr, NA_N, 256, 256, MODE_SCORE, q2, scores + 1);
    softmax2<<<1, 1, 0, stream>>>(scores, 1.f / (float)NA_N, attnw);
    combine2_v4<<<dim3((unsigned)(((size_t)NA_N * 64 + 255) / 256)), 256, 0, stream>>>(
        (const float4*)oa1, (const float4*)oa2, attnw, (float4*)d_out, (size_t)NA_N * 64);

    (void)in_sizes; (void)n_in; (void)out_size; (void)ws_size;
}
